// HAM_74302934221523
// MI455X (gfx1250) — matrix operand order repaired and checked by reading, not compiled, not run on hardware
//
#include <hip/hip_runtime.h>

// Problem sizes (compile-time constants from the reference)
#define B_    64
#define C_    512
#define S_    768
#define K_    128
#define KN_   256
#define A_    200
#define H_    1024
#define TWO_S 1536
#define BN_EPS 1e-5f

typedef __attribute__((ext_vector_type(16))) __bf16 v16bf;
typedef __attribute__((ext_vector_type(8)))  float  v8f;

// A-fragment: 16x32 bf16. ISA 16-bit A 16x32 layout:
// lanes 0-15 hold K = {0..7, 16..23}, lanes 16-31 hold K = {8..15, 24..31} of row M=lane&15.
// Source rows are K-contiguous f32; chunks are 32B aligned -> b128 loads + native bf16 cvt.
__device__ __forceinline__ v16bf load_a16x32(const float* __restrict__ p, int ld, int lane) {
  int m = lane & 15, hi = lane >> 4;
  const float* r = p + (size_t)m * ld + hi * 8;
  float f[16];
  *(float4*)(f + 0)  = ((const float4*)r)[0];
  *(float4*)(f + 4)  = ((const float4*)r)[1];
  *(float4*)(f + 8)  = ((const float4*)(r + 16))[0];
  *(float4*)(f + 12) = ((const float4*)(r + 16))[1];
  v16bf a;
#pragma unroll
  for (int i = 0; i < 16; ++i) a[i] = (__bf16)f[i];
  return a;
}

// B-fragment: 32x16 bf16. Column N = lane&15; lanes 0-15 hold K={0..7, 16..23},
// lanes 16-31 hold K={8..15, 24..31}. NT source: column n is contiguous memory row n.
__device__ __forceinline__ v16bf load_b32x16(const float* __restrict__ p, int ld, int lane) {
  int n = lane & 15, hi = lane >> 4;
  const float* r = p + (size_t)n * ld + hi * 8;
  float f[16];
  *(float4*)(f + 0)  = ((const float4*)r)[0];
  *(float4*)(f + 4)  = ((const float4*)r)[1];
  *(float4*)(f + 8)  = ((const float4*)(r + 16))[0];
  *(float4*)(f + 12) = ((const float4*)(r + 16))[1];
  v16bf b;
#pragma unroll
  for (int i = 0; i < 16; ++i) b[i] = (__bf16)f[i];
  return b;
}

// ---------------- Kernel 1: partial column sums of x over C ----------------
// x: [B, C, S]. Grid (B, 4 c-chunks), block 192: thread owns 4 consecutive s
// (float4 / global_load_b128, coalesced 3KB lines per c-row). Dominant 100.7 MB stream.
__global__ void colsum_partial(const float* __restrict__ x, float* __restrict__ sxp) {
  int b  = blockIdx.x;
  int cc = blockIdx.y;                        // c chunk: 128 rows each
  int s4 = threadIdx.x * 4;                   // 192 threads * 4 = 768 = S
  const float* px = x + ((size_t)b * C_ + (size_t)cc * 128) * S_ + s4;
  float4 acc = {0.f, 0.f, 0.f, 0.f};
#pragma unroll 8
  for (int c = 0; c < 128; ++c) {
    if ((c & 7) == 0) __builtin_prefetch(px + 8 * S_, 0, 0);  // global_prefetch_b8
    float4 v = *(const float4*)px;
    acc.x += v.x; acc.y += v.y; acc.z += v.z; acc.w += v.w;
    px += S_;
  }
  *(float4*)(sxp + ((size_t)(b * 4 + cc)) * S_ + s4) = acc;
}

// ------- Kernel 2: reduce partials, build local_input = [sum/C ; sum/K] -------
__global__ void finalize_li(const float* __restrict__ sxp, float* __restrict__ li) {
  int b  = blockIdx.x;
  int s4 = threadIdx.x * 4;
  float4 sum = {0.f, 0.f, 0.f, 0.f};
#pragma unroll
  for (int j = 0; j < 4; ++j) {
    float4 v = *(const float4*)(sxp + ((size_t)(b * 4 + j)) * S_ + s4);
    sum.x += v.x; sum.y += v.y; sum.z += v.z; sum.w += v.w;
  }
  float4 xm = {sum.x * (1.f / C_), sum.y * (1.f / C_), sum.z * (1.f / C_), sum.w * (1.f / C_)};
  float4 av = {sum.x * (1.f / K_), sum.y * (1.f / K_), sum.z * (1.f / K_), sum.w * (1.f / K_)};
  *(float4*)(li + (size_t)b * TWO_S + s4)      = xm;  // x_mean
  *(float4*)(li + (size_t)b * TWO_S + S_ + s4) = av;  // attn_out_avg (softmax collapse)
}

// ------------- Kernel 3: fc = local_input @ W_t^T + b_t  (WMMA) -------------
// M=64 (4 tiles), N=1024, K=1536. Block = 8 waves: wave = (m_tile[0..3], n_half[0..1]).
__global__ void fc_gemm(const float* __restrict__ li, const float* __restrict__ Wt,
                        const float* __restrict__ bt, float* __restrict__ fc) {
  int wave = threadIdx.x >> 5, lane = threadIdx.x & 31;
  int m0 = (wave & 3) * 16;
  int n0 = (blockIdx.x * 2 + (wave >> 2)) * 16;
  v8f acc = {0.f, 0.f, 0.f, 0.f, 0.f, 0.f, 0.f, 0.f};
  for (int kk = 0; kk < TWO_S; kk += 32) {
    v16bf a = load_a16x32(li + (size_t)m0 * TWO_S + kk, TWO_S, lane);
    v16bf b = load_b32x16(Wt + (size_t)n0 * TWO_S + kk, TWO_S, lane);
    acc = __builtin_amdgcn_wmma_f32_16x16x32_bf16(false, a, false, b,
                                                  (short)0, acc, false, false);
  }
  int n = lane & 15, hi = lane >> 4;
  int h = n0 + n;
  float bias = bt[h];
#pragma unroll
  for (int r = 0; r < 8; ++r) {
    int bb = m0 + r + hi * 8;                // C/D layout: VGPR r -> M = r (+8 for hi lanes)
    fc[(size_t)bb * H_ + h] = acc[r] + bias;
  }
}

// --------- Kernel 4: BatchNorm(train, biased var) + ReLU -> output 2 ---------
__global__ void bn_relu(const float* __restrict__ fc, const float* __restrict__ gamma,
                        const float* __restrict__ beta, float* __restrict__ fcout) {
  int h = blockIdx.x * 256 + threadIdx.x;
  float sum = 0.f;
  for (int b = 0; b < B_; ++b) sum += fc[(size_t)b * H_ + h];
  float mu = sum * (1.f / B_);
  float vs = 0.f;
  for (int b = 0; b < B_; ++b) { float d = fc[(size_t)b * H_ + h] - mu; vs += d * d; }
  float rstd = rsqrtf(vs * (1.f / B_) + BN_EPS);
  float g = gamma[h], be = beta[h];
  for (int b = 0; b < B_; ++b) {
    float v = (fc[(size_t)b * H_ + h] - mu) * rstd * g + be;
    fcout[(size_t)b * H_ + h] = v > 0.f ? v : 0.f;
  }
}

// ------ Kernel 5: local_score = sigmoid(fc_out @ W_l^T + b_l)  (WMMA) ------
// M=64, N=128, K=1024. grid=4 m-tiles, 8 waves = 8 n-tiles.
__global__ void score_gemm(const float* __restrict__ fcout, const float* __restrict__ Wl,
                           const float* __restrict__ bl, float* __restrict__ score) {
  int wave = threadIdx.x >> 5, lane = threadIdx.x & 31;
  int m0 = blockIdx.x * 16;
  int n0 = wave * 16;
  v8f acc = {0.f, 0.f, 0.f, 0.f, 0.f, 0.f, 0.f, 0.f};
  for (int kk = 0; kk < H_; kk += 32) {
    v16bf a = load_a16x32(fcout + (size_t)m0 * H_ + kk, H_, lane);
    v16bf b = load_b32x16(Wl + (size_t)n0 * H_ + kk, H_, lane);
    acc = __builtin_amdgcn_wmma_f32_16x16x32_bf16(false, a, false, b,
                                                  (short)0, acc, false, false);
  }
  int n = lane & 15, hi = lane >> 4;
  int k = n0 + n;
  float bias = bl[k];
#pragma unroll
  for (int r = 0; r < 8; ++r) {
    int bb = m0 + r + hi * 8;
    float v = acc[r] + bias;
    score[(size_t)bb * K_ + k] = 1.0f / (1.0f + __expf(-v));
  }
}

// ---------- Kernel 6: Q_h = score @ G_h, broadcast over A -> output 3 ----------
__global__ void qh_broadcast(const float* __restrict__ score, const float* __restrict__ Gh,
                             float* __restrict__ omega) {
  int b = blockIdx.x;
  int n = threadIdx.x;                        // 0..255 (K_NEXT)
  float q = 0.f;
  for (int k = 0; k < K_; ++k) q += score[(size_t)b * K_ + k] * Gh[(size_t)k * KN_ + n];
  float4 qv = {q, q, q, q};
  float4* dst = (float4*)(omega + ((size_t)b * KN_ + n) * A_);  // 200*4B = 16B-aligned
#pragma unroll 5
  for (int a = 0; a < A_ / 4; ++a) dst[a] = qv;                 // b128 stores, 13.1 MB total
}

extern "C" void kernel_launch(void* const* d_in, const int* in_sizes, int n_in,
                              void* d_out, int out_size, void* d_ws, size_t ws_size,
                              hipStream_t stream) {
  (void)in_sizes; (void)n_in; (void)out_size; (void)ws_size;
  const float* x     = (const float*)d_in[0];
  // d_in[1] omega_h, d_in[2] W_s1, d_in[3] W_s2: dead after softmax-mean collapse
  const float* Wt    = (const float*)d_in[4];
  const float* bt    = (const float*)d_in[5];
  const float* gamma = (const float*)d_in[6];
  const float* beta  = (const float*)d_in[7];
  const float* Wl    = (const float*)d_in[8];
  const float* bl    = (const float*)d_in[9];
  const float* Gh    = (const float*)d_in[10];

  float* out   = (float*)d_out;
  float* score = out;                        // [64,128]   = 8192
  float* fcout = out + 8192;                 // [64,1024]  = 65536
  float* omega = out + 73728;                // [64,256,200]

  float* ws  = (float*)d_ws;
  float* sxp = ws;                           // [64,4,768] = 196608 floats
  float* li  = ws + 196608;                  // [64,1536]  = 98304 floats
  float* fc  = ws + 294912;                  // [64,1024]  = 65536 floats

  colsum_partial<<<dim3(B_, 4), 192, 0, stream>>>(x, sxp);
  finalize_li  <<<B_,           192, 0, stream>>>(sxp, li);
  fc_gemm      <<<H_ / 32,      256, 0, stream>>>(li, Wt, bt, fc);
  bn_relu      <<<H_ / 256,     256, 0, stream>>>(fc, gamma, beta, fcout);
  score_gemm   <<<B_ / 16,      256, 0, stream>>>(fcout, Wl, bl, score);
  qh_broadcast <<<B_,           256, 0, stream>>>(score, Gh, omega);
}
